// NSGAT_37203006718152
// MI455X (gfx1250) — matrix operand order repaired and checked by reading, not compiled, not run on hardware
//
#include <hip/hip_runtime.h>
#include <hip/hip_bf16.h>

typedef __attribute__((ext_vector_type(16))) __bf16 v16bf;
typedef __attribute__((ext_vector_type(8)))  float  v8f;

// ---------------- helpers ----------------
__device__ __forceinline__ unsigned short f32_to_bf16_rne(float f) {
    unsigned u = __float_as_uint(f);
    u += 0x7FFFu + ((u >> 16) & 1u);
    return (unsigned short)(u >> 16);
}
// order-preserving float -> uint key (for atomicMax on floats)
__device__ __forceinline__ unsigned fkey(float f) {
    unsigned u = __float_as_uint(f);
    return (u & 0x80000000u) ? ~u : (u | 0x80000000u);
}
__device__ __forceinline__ float funkey(unsigned k) {
    unsigned u = (k & 0x80000000u) ? (k & 0x7FFFFFFFu) : ~k;
    return __uint_as_float(u);
}

// ---------------- conversions ----------------
__global__ void cvt_f32_bf16(const float* __restrict__ in,
                             unsigned short* __restrict__ out, int n) {
    int i = blockIdx.x * blockDim.x + threadIdx.x;
    if (i < n) out[i] = f32_to_bf16_rne(in[i]);
}
__global__ void cvt_pad_bf16(const float* __restrict__ in,
                             unsigned short* __restrict__ out,
                             int rows, int cin, int cout) {
    int i = blockIdx.x * blockDim.x + threadIdx.x;
    if (i >= rows * cout) return;
    int r = i / cout, c = i % cout;
    out[i] = (c < cin) ? f32_to_bf16_rne(in[r * cin + c]) : (unsigned short)0;
}

// ---------------- bf16 WMMA GEMM: C[MxN] = A[MxK] * B[KxN] ----------------
// block tile 128x64, 128 threads = 4 waves; wave w -> rows [w*32, w*32+32)
// (two 16x16x32 WMMA row-tiles per wave x 4 col-tiles = 8 WMMAs per K-step)
__global__ __launch_bounds__(128) void gemm_bf16_wmma(
    const unsigned short* __restrict__ A,   // M x K bf16
    const unsigned short* __restrict__ B,   // K x N bf16
    float* __restrict__ C,                  // M x N f32
    int M, int N, int K)
{
    // As[row][k]        : ushort idx = row*32 + k          (128 x 32)
    // Bs fragment-major : ushort idx = col*32 + kp*2 + par (col 0..63, kp 0..15)
    //   element B[k][col] -> kp = k>>1, par = k&1
    __shared__ unsigned short As[128 * 32];
    __shared__ unsigned short Bs[64 * 32];

    const int tid  = threadIdx.x;
    const int wave = tid >> 5;
    const int lane = tid & 31;
    const int l15  = lane & 15;
    const int hi   = lane >> 4;
    const int m0   = blockIdx.x * 128;
    const int n0   = blockIdx.y * 64;

    v8f acc[2][4] = {};

    // per-thread staging coordinates (loop-invariant)
    const int a_gr = m0 + tid;                 // A: one row per thread
    const int b_r  = tid >> 2;                 // B: local k row 0..31
    const int b_c0 = (tid & 3) * 16;           // B: col base
    const int b_kp = b_r >> 1, b_par = b_r & 1;

    for (int k0 = 0; k0 < K; k0 += 32) {
        // ---- stage A tile: 128 rows x 32 cols ----
        {
            uint4 d0 = make_uint4(0, 0, 0, 0), d1 = d0;
            if (a_gr < M) {
                const uint4* p = (const uint4*)(A + (size_t)a_gr * K + k0);
                d0 = p[0]; d1 = p[1];
                if (k0 + 32 < K)   // prefetch next K-step (-> global_prefetch_b8)
                    __builtin_prefetch(A + (size_t)a_gr * K + k0 + 32, 0, 1);
            }
            uint4* s = (uint4*)(As + tid * 32);
            s[0] = d0; s[1] = d1;
        }
        // ---- stage B tile: 32 rows x 64 cols -> fragment-major layout ----
        {
            const uint4* p = (const uint4*)(B + (size_t)(k0 + b_r) * N + n0 + b_c0);
            uint4 t0 = p[0], t1 = p[1];
            if (k0 + 32 < K)
                __builtin_prefetch(B + (size_t)(k0 + 32 + b_r) * N + n0 + b_c0, 0, 1);
            unsigned short tmp[16];
            *(uint4*)&tmp[0] = t0; *(uint4*)&tmp[8] = t1;
#pragma unroll
            for (int j = 0; j < 16; ++j)
                Bs[(b_c0 + j) * 32 + b_kp * 2 + b_par] = tmp[j];
        }
        __syncthreads();

        // ---- A fragments (ISA 16-bit 16x32 layout): k = 8*hi+s, then 16+8*hi+s per lane ----
        union Frag { uint4 q[2]; v16bf v; };
        Frag af[2];
#pragma unroll
        for (int i = 0; i < 2; ++i) {
            const unsigned short* ap = As + (wave * 32 + i * 16 + l15) * 32 + hi * 8;
            af[i].q[0] = *(const uint4*)ap;
            af[i].q[1] = *(const uint4*)(ap + 16);
        }
        // ---- 4 col-tiles x 2 row-tiles of WMMA ----
#pragma unroll
        for (int nt = 0; nt < 4; ++nt) {
            Frag bfrag;
            const unsigned short* bp = Bs + (nt * 16 + l15) * 32 + hi * 8;
            bfrag.q[0] = *(const uint4*)bp;
            bfrag.q[1] = *(const uint4*)(bp + 16);
#pragma unroll
            for (int i = 0; i < 2; ++i) {
                acc[i][nt] = __builtin_amdgcn_wmma_f32_16x16x32_bf16(
                    false, af[i].v, false, bfrag.v, (short)0, acc[i][nt], false, false);
            }
        }
        __syncthreads();
    }

    // ---- epilogue: C/D layout M = v + 8*hi, N = lane&15 ----
    if (m0 + 128 <= M) {
        // fast path: full tile, no guards
        float* cbase = C + (size_t)(m0 + wave * 32 + 8 * hi) * N + n0 + l15;
#pragma unroll
        for (int i = 0; i < 2; ++i) {
#pragma unroll
            for (int v = 0; v < 8; ++v) {
                float* cp = cbase + (size_t)(i * 16 + v) * N;
#pragma unroll
                for (int nt = 0; nt < 4; ++nt)
                    cp[nt * 16] = acc[i][nt][v];
            }
        }
    } else {
#pragma unroll
        for (int i = 0; i < 2; ++i) {
#pragma unroll
            for (int v = 0; v < 8; ++v) {
                int m = m0 + wave * 32 + i * 16 + v + 8 * hi;
                if (m < M) {
                    float* cp = C + (size_t)m * N + n0 + l15;
#pragma unroll
                    for (int nt = 0; nt < 4; ++nt)
                        cp[nt * 16] = acc[i][nt][v];
                }
            }
        }
    }
}

// ---------------- attention pieces ----------------
__global__ void el_er_kernel(const float* __restrict__ feat,
                             const float* __restrict__ al,
                             const float* __restrict__ ar,
                             float* __restrict__ el, float* __restrict__ er,
                             int Nsrc, int ndst, int H, int F, int ldf)
{
    int i = blockIdx.x * blockDim.x + threadIdx.x;
    if (i >= Nsrc * H) return;
    int node = i / H, h = i % H;
    const float* fp = feat + (size_t)node * ldf + h * F;
    float sl = 0.f, sr = 0.f;
    for (int f = 0; f < F; ++f) { float v = fp[f]; sl += v * al[h * F + f]; sr += v * ar[h * F + f]; }
    el[i] = sl;
    if (node < ndst) er[node * H + h] = sr;
}

__global__ void edge_att_kernel(const int* __restrict__ src, const int* __restrict__ dst,
                                const float* __restrict__ el, const float* __restrict__ er,
                                float* __restrict__ ebuf, unsigned* __restrict__ emaxk,
                                int E, int H)
{
    int i = blockIdx.x * blockDim.x + threadIdx.x;
    if (i >= E * H) return;
    int e = i / H, h = i % H;
    float v = el[src[e] * H + h] + er[dst[e] * H + h];
    v = (v > 0.f) ? v : 0.2f * v;                 // LeakyReLU(0.2)
    ebuf[i] = v;
    atomicMax(&emaxk[dst[e] * H + h], fkey(v));
}

__global__ void edge_exp_kernel(const int* __restrict__ dst,
                                const unsigned* __restrict__ emaxk,
                                float* __restrict__ ebuf, float* __restrict__ denom,
                                int E, int H)
{
    int i = blockIdx.x * blockDim.x + threadIdx.x;
    if (i >= E * H) return;
    int e = i / H, h = i % H;
    float m = funkey(emaxk[dst[e] * H + h]);
    float a = expf(ebuf[i] - m);
    ebuf[i] = a;
    atomicAdd(&denom[dst[e] * H + h], a);
}

template <int VEC>
__global__ void edge_aggr_kernel(const int* __restrict__ src, const int* __restrict__ dst,
                                 const float* __restrict__ ebuf, const float* __restrict__ denom,
                                 const float* __restrict__ feat, float* __restrict__ rst,
                                 int E, int H, int F, int ldf)
{
    long long i = (long long)blockIdx.x * blockDim.x + threadIdx.x;
    const int HF  = H * F;
    const int HFv = HF / VEC;
    if (i >= (long long)E * HFv) return;
    int e  = (int)(i / HFv);
    int cv = (int)(i % HFv) * VEC;          // VEC-aligned channel (stays within one head)
    int h  = cv / F;
    int s  = src[e], d = dst[e];
    float alpha = ebuf[e * H + h] / denom[d * H + h];
    const float* fp = feat + (size_t)s * ldf + cv;
    float*       rp = rst  + (size_t)d * ldf + cv;
#pragma unroll
    for (int j = 0; j < VEC; ++j)
        atomicAdd(rp + j, alpha * fp[j]);
}

__global__ void bias_bf16_kernel(const float* __restrict__ rst, const float* __restrict__ b,
                                 unsigned short* __restrict__ out, int n, int cols)
{
    int i = blockIdx.x * blockDim.x + threadIdx.x;
    if (i >= n * cols) return;
    out[i] = f32_to_bf16_rne(rst[i] + b[i % cols]);
}

__global__ void final_kernel(const float* __restrict__ rst, const float* __restrict__ b,
                             float* __restrict__ out, int n, int F, int ldf)
{
    int i = blockIdx.x * blockDim.x + threadIdx.x;
    if (i >= n * F) return;
    int r = i / F, f = i % F;
    out[i] = rst[(size_t)r * ldf + f] + b[f];
}

// ---------------- launch ----------------
extern "C" void kernel_launch(void* const* d_in, const int* in_sizes, int n_in,
                              void* d_out, int out_size, void* d_ws, size_t ws_size,
                              hipStream_t stream) {
    (void)in_sizes; (void)n_in; (void)out_size; (void)ws_size;
    const float* x   = (const float*)d_in[0];
    const float* W[3]  = {(const float*)d_in[1], (const float*)d_in[5], (const float*)d_in[9]};
    const float* al[3] = {(const float*)d_in[2], (const float*)d_in[6], (const float*)d_in[10]};
    const float* ar[3] = {(const float*)d_in[3], (const float*)d_in[7], (const float*)d_in[11]};
    const float* bb[3] = {(const float*)d_in[4], (const float*)d_in[8], (const float*)d_in[12]};
    const int* src[3]  = {(const int*)d_in[13], (const int*)d_in[15], (const int*)d_in[17]};
    const int* dstp[3] = {(const int*)d_in[14], (const int*)d_in[16], (const int*)d_in[18]};

    const int Nsrc[3] = {120000, 30000, 8000};
    const int Ndst[3] = {30000, 8000, 2048};
    const int Ecnt[3] = {480000, 160000, 40000};
    const int Hh[3]   = {4, 4, 1};
    const int Ff[3]   = {64, 64, 47};
    const int Npad[3] = {256, 256, 64};   // GEMM N / feat & rst leading dim

    // workspace carving
    char* ws = (char*)d_ws;
    size_t off = 0;
    auto carve = [&](size_t bytes) { size_t p = off; off += (bytes + 255) & ~(size_t)255; return (void*)(ws + p); };
    unsigned short* XB   = (unsigned short*)carve((size_t)120000 * 256 * 2);
    unsigned short* WB   = (unsigned short*)carve((size_t)256 * 256 * 2);
    float*    FEAT  = (float*)carve((size_t)120000 * 256 * 4);
    float*    EL    = (float*)carve((size_t)120000 * 4 * 4);
    float*    ER    = (float*)carve((size_t)30000 * 4 * 4);
    float*    EBUF  = (float*)carve((size_t)480000 * 4 * 4);
    unsigned* EMAX  = (unsigned*)carve((size_t)30000 * 4 * 4);
    float*    DEN   = (float*)carve((size_t)30000 * 4 * 4);
    float*    RST   = (float*)carve((size_t)30000 * 256 * 4);

    const int BT = 256;
    // input -> bf16
    {
        int n = 120000 * 256;
        cvt_f32_bf16<<<(n + BT - 1) / BT, BT, 0, stream>>>(x, XB, n);
    }

    for (int L = 0; L < 3; ++L) {
        int M = Nsrc[L], nd = Ndst[L], E = Ecnt[L], H = Hh[L], F = Ff[L], NP = Npad[L];
        // weights -> bf16 (layer 2 zero-padded 47 -> 64)
        if (L < 2) {
            int n = 256 * 256;
            cvt_f32_bf16<<<(n + BT - 1) / BT, BT, 0, stream>>>(W[L], WB, n);
        } else {
            int n = 256 * 64;
            cvt_pad_bf16<<<(n + BT - 1) / BT, BT, 0, stream>>>(W[L], WB, 256, 47, 64);
        }
        // feat = h @ W  (bf16 WMMA, f32 out)
        {
            dim3 grid((M + 127) / 128, NP / 64);
            gemm_bf16_wmma<<<grid, 128, 0, stream>>>(XB, WB, FEAT, M, NP, 256);
        }
        // el / er
        {
            int n = M * H;
            el_er_kernel<<<(n + BT - 1) / BT, BT, 0, stream>>>(FEAT, al[L], ar[L], EL, ER, M, nd, H, F, NP);
        }
        // zero per-dst accumulators
        hipMemsetAsync(EMAX, 0, (size_t)nd * H * 4, stream);
        hipMemsetAsync(DEN,  0, (size_t)nd * H * 4, stream);
        hipMemsetAsync(RST,  0, (size_t)nd * NP * 4, stream);
        // pass 1: leaky_relu + segment max
        {
            int n = E * H;
            edge_att_kernel<<<(n + BT - 1) / BT, BT, 0, stream>>>(src[L], dstp[L], EL, ER, EBUF, EMAX, E, H);
        }
        // pass 2: exp + segment sum
        {
            int n = E * H;
            edge_exp_kernel<<<(n + BT - 1) / BT, BT, 0, stream>>>(dstp[L], EMAX, EBUF, DEN, E, H);
        }
        // pass 3: alpha-weighted aggregation (VEC=4 when HF divisible by 4)
        if (L < 2) {
            long long n = (long long)E * H * F / 4;
            int blocks = (int)((n + BT - 1) / BT);
            edge_aggr_kernel<4><<<blocks, BT, 0, stream>>>(src[L], dstp[L], EBUF, DEN, FEAT, RST, E, H, F, NP);
        } else {
            long long n = (long long)E * H * F;
            int blocks = (int)((n + BT - 1) / BT);
            edge_aggr_kernel<1><<<blocks, BT, 0, stream>>>(src[L], dstp[L], EBUF, DEN, FEAT, RST, E, H, F, NP);
        }
        if (L < 2) {
            int n = nd * 256;
            bias_bf16_kernel<<<(n + BT - 1) / BT, BT, 0, stream>>>(RST, bb[L], XB, nd, 256);
        } else {
            int n = nd * 47;
            final_kernel<<<(n + BT - 1) / BT, BT, 0, stream>>>(RST, bb[L], (float*)d_out, nd, 47, 64);
        }
    }
}
